// Decoder_5531917877559
// MI455X (gfx1250) — compile-verified, matrix operand order checked by reading, not run on hardware
//
#include <hip/hip_runtime.h>
#include <hip/hip_bf16.h>

#define B_     64
#define T_ENC_ 256
#define T_DEC_ 200
#define MEL_   80
#define PRE_   256
#define LSTM_  1024
#define EMB_   768
#define ATTN_  128
#define FILT_  32
#define KS_    31

typedef __attribute__((ext_vector_type(16))) __bf16 bf16x16;
typedef __attribute__((ext_vector_type(8)))  __bf16 bf16x8;
typedef __attribute__((ext_vector_type(8)))  float  f32x8;

// ---------------------------------------------------------------------------
// WMMA fragment helpers (CDNA5 bf16 16x16x32 layouts, wave32)
//   A (16xK panel, row-major, ld elems/row):
//     lanes 0-15 : row = lane, K = k0+0..7  and k0+16..23
//     lanes 16-31: row = lane-16, K = k0+8..15 and k0+24..31
//   B operand stored as W[n][k] row-major (i.e. B^T) uses the same pattern
//   with "row" = output column n.
// ---------------------------------------------------------------------------
__device__ __forceinline__ bf16x16 load_frag_bf16(const __bf16* panel, int ld,
                                                  int k0, int lane) {
  int r  = lane & 15;
  int ko = (lane < 16) ? 0 : 8;
  const __bf16* p = panel + (size_t)r * (size_t)ld + k0 + ko;
  bf16x8 lo = *(const bf16x8*)(p);
  bf16x8 hi = *(const bf16x8*)(p + 16);
  return __builtin_shufflevector(lo, hi, 0,1,2,3,4,5,6,7,8,9,10,11,12,13,14,15);
}

// C/D 16x16 f32 tile: lane n = lane&15 ; VGPR r -> row r (+8 for lanes>=16)
__device__ __forceinline__ void store_tile_f32(float* base, int ld, int lane,
                                               f32x8 c) {
  int n  = lane & 15;
  int mo = (lane < 16) ? 0 : 8;
#pragma unroll
  for (int r = 0; r < 8; ++r) base[(size_t)(mo + r) * ld + n] = c[r];
}

__device__ __forceinline__ float sigmoidf_(float x) {
  return 1.0f / (1.0f + __expf(-x));
}

// ---------------------------------------------------------------------------
// Generic bf16 WMMA GEMM: C[MxN] = A[MxK] * W[NxK]^T, optional ReLU.
// grid = (N/64, M/64), block = 128 (4 waves); wave w -> 16-row M-tile.
// Software-pipelined: next K-step fragments are prefetched before the current
// step's WMMAs so the compiler can overlap loads with matrix issue.
// ---------------------------------------------------------------------------
__global__ void wmma_gemm_bf16(const __bf16* __restrict__ A,
                               const __bf16* __restrict__ W,
                               float* __restrict__ C,
                               int M, int N, int K, int relu) {
  int lane = threadIdx.x & 31;
  int wave = threadIdx.x >> 5;
  int m0 = blockIdx.y * 64 + wave * 16;
  int n0 = blockIdx.x * 64;
  if (m0 >= M) return;
  f32x8 acc[4] = {f32x8{}, f32x8{}, f32x8{}, f32x8{}};
  const __bf16* Ap = A + (size_t)m0 * K;

  bf16x16 a_cur = load_frag_bf16(Ap, K, 0, lane);
  bf16x16 b_cur[4];
#pragma unroll
  for (int j = 0; j < 4; ++j)
    b_cur[j] = load_frag_bf16(W + (size_t)(n0 + 16 * j) * K, K, 0, lane);

  for (int k0 = 0; k0 < K; k0 += 32) {
    int kn = (k0 + 32 < K) ? (k0 + 32) : k0;  // clamp: last iter re-loads
    bf16x16 a_nxt = load_frag_bf16(Ap, K, kn, lane);
    bf16x16 b_nxt[4];
#pragma unroll
    for (int j = 0; j < 4; ++j)
      b_nxt[j] = load_frag_bf16(W + (size_t)(n0 + 16 * j) * K, K, kn, lane);
#pragma unroll
    for (int j = 0; j < 4; ++j)
      acc[j] = __builtin_amdgcn_wmma_f32_16x16x32_bf16(
          false, a_cur, false, b_cur[j], (short)0, acc[j], false, false);
    a_cur = a_nxt;
#pragma unroll
    for (int j = 0; j < 4; ++j) b_cur[j] = b_nxt[j];
  }
#pragma unroll
  for (int j = 0; j < 4; ++j) {
    f32x8 c = acc[j];
    if (relu) {
#pragma unroll
      for (int r = 0; r < 8; ++r) c[r] = fmaxf(c[r], 0.0f);
    }
    store_tile_f32(C + (size_t)m0 * N + n0 + 16 * j, N, lane, c);
  }
}

// ---------------------------------------------------------------------------
// Fused LSTM cell: gates = A[64x2048] * Wcat[4096x2048]^T (+bias), then
// pointwise c/h update, all in-registers. A = [x_t | ctx | h_prev] (bf16).
// Wcat rows: gate g occupies rows [g*1024, (g+1)*1024).  Gate order: i,f,g,o.
// grid = 64 blocks (16 hidden cols each), block = 128 (4 waves = 4 M-tiles).
// Software-pipelined like wmma_gemm_bf16.
// ---------------------------------------------------------------------------
__global__ void lstm_fused_kernel(const __bf16* __restrict__ A,
                                  const __bf16* __restrict__ Wcat,
                                  const float* __restrict__ bias,
                                  float* __restrict__ h,
                                  float* __restrict__ c) {
  const int KTOT = 2 * LSTM_;
  int lane = threadIdx.x & 31;
  int wave = threadIdx.x >> 5;
  int m0 = wave * 16;            // batch tile
  int n0 = blockIdx.x * 16;      // hidden-column tile
  f32x8 acc[4] = {f32x8{}, f32x8{}, f32x8{}, f32x8{}};
  const __bf16* Ap = A + (size_t)m0 * KTOT;

  bf16x16 a_cur = load_frag_bf16(Ap, KTOT, 0, lane);
  bf16x16 b_cur[4];
#pragma unroll
  for (int g = 0; g < 4; ++g)
    b_cur[g] =
        load_frag_bf16(Wcat + (size_t)(g * LSTM_ + n0) * KTOT, KTOT, 0, lane);

  for (int k0 = 0; k0 < KTOT; k0 += 32) {
    int kn = (k0 + 32 < KTOT) ? (k0 + 32) : k0;
    bf16x16 a_nxt = load_frag_bf16(Ap, KTOT, kn, lane);
    bf16x16 b_nxt[4];
#pragma unroll
    for (int g = 0; g < 4; ++g)
      b_nxt[g] = load_frag_bf16(Wcat + (size_t)(g * LSTM_ + n0) * KTOT, KTOT,
                                kn, lane);
#pragma unroll
    for (int g = 0; g < 4; ++g)
      acc[g] = __builtin_amdgcn_wmma_f32_16x16x32_bf16(
          false, a_cur, false, b_cur[g], (short)0, acc[g], false, false);
    a_cur = a_nxt;
#pragma unroll
    for (int g = 0; g < 4; ++g) b_cur[g] = b_nxt[g];
  }

  int n  = lane & 15;
  int mo = (lane < 16) ? 0 : 8;
#pragma unroll
  for (int r = 0; r < 8; ++r) {
    int m   = m0 + mo + r;
    int col = n0 + n;
    float gi = acc[0][r] + bias[col];
    float gf = acc[1][r] + bias[LSTM_ + col];
    float gg = acc[2][r] + bias[2 * LSTM_ + col];
    float go = acc[3][r] + bias[3 * LSTM_ + col];
    float cold = c[(size_t)m * LSTM_ + col];
    float c2 = sigmoidf_(gf) * cold + sigmoidf_(gi) * tanhf(gg);
    float h2 = sigmoidf_(go) * tanhf(c2);
    c[(size_t)m * LSTM_ + col] = c2;
    h[(size_t)m * LSTM_ + col] = h2;
  }
}

// ---------------------------------------------------------------------------
// Small utility kernels
// ---------------------------------------------------------------------------
__global__ void cvt_f32_to_bf16(const float* __restrict__ in,
                                __bf16* __restrict__ out, size_t n) {
  size_t i = (size_t)blockIdx.x * blockDim.x + threadIdx.x;
  if (i < n) out[i] = (__bf16)in[i];
}

__global__ void cat_weights_bf16(const float* __restrict__ wih,
                                 const float* __restrict__ whh,
                                 __bf16* __restrict__ out) {
  size_t i = (size_t)blockIdx.x * blockDim.x + threadIdx.x;
  if (i >= (size_t)(4 * LSTM_) * (2 * LSTM_)) return;
  int k = (int)(i & (2 * LSTM_ - 1));
  size_t row = i >> 11;
  float v = (k < LSTM_) ? wih[row * LSTM_ + k] : whh[row * LSTM_ + (k - LSTM_)];
  out[i] = (__bf16)v;
}

__global__ void bias_add_kernel(const float* __restrict__ a,
                                const float* __restrict__ b,
                                float* __restrict__ out, int n) {
  int i = blockIdx.x * blockDim.x + threadIdx.x;
  if (i < n) out[i] = a[i] + b[i];
}

__global__ void mul_vec_kernel(const float* __restrict__ a,
                               const float* __restrict__ b,
                               float* __restrict__ out, int n) {
  int i = blockIdx.x * blockDim.x + threadIdx.x;
  if (i < n) out[i] = a[i] * b[i];
}

__global__ void zero_f32_kernel(float* __restrict__ p, size_t n) {
  size_t i = (size_t)blockIdx.x * blockDim.x + threadIdx.x;
  if (i < n) p[i] = 0.0f;
}

// Prenet layer 1 (K=80, VALU): rows ordered (t*64+b); t==0 -> go frame (zeros)
__global__ void prenet1_kernel(const float* __restrict__ dec_in,
                               const float* __restrict__ w1,
                               float* __restrict__ h1) {
  int idx = blockIdx.x * blockDim.x + threadIdx.x;  // T_DEC*B*PRE
  if (idx >= T_DEC_ * B_ * PRE_) return;
  int row = idx >> 8;
  int p   = idx & 255;
  int t = row >> 6;
  int b = row & 63;
  float s = 0.0f;
  if (t > 0) {
    const float* x = dec_in + ((size_t)b * T_DEC_ + (t - 1)) * MEL_;
    const float* w = w1 + (size_t)p * MEL_;
#pragma unroll 8
    for (int m = 0; m < MEL_; ++m) s += w[m] * x[m];
  }
  h1[idx] = fmaxf(s, 0.0f);
}

// Build A = [x_t(256) | ctx(768) | h(1024)] in bf16 for one LSTM cell
__global__ void pack_A_kernel(const float* __restrict__ xs_t,
                              const float* __restrict__ ctx,
                              const float* __restrict__ h,
                              __bf16* __restrict__ A) {
  int idx = blockIdx.x * blockDim.x + threadIdx.x;  // 64*2048
  if (idx >= B_ * 2 * LSTM_) return;
  int b = idx >> 11;
  int k = idx & 2047;
  float v;
  if (k < PRE_)        v = xs_t[(size_t)b * PRE_ + k];
  else if (k < LSTM_)  v = ctx[(size_t)b * EMB_ + (k - PRE_)];
  else                 v = h[(size_t)b * LSTM_ + (k - LSTM_)];
  A[idx] = (__bf16)v;
}

// Location conv + dense + tanh energy reduce. One block per batch b.
__global__ void energies_kernel(const float* __restrict__ aw,
                                const float* __restrict__ awc,
                                const float* __restrict__ pq,
                                const float* __restrict__ pm,
                                const float* __restrict__ convw,
                                const float* __restrict__ densew,
                                const float* __restrict__ vv,
                                const float* __restrict__ abias,
                                float* __restrict__ energies) {
  int b   = blockIdx.x;
  int tid = threadIdx.x;
  __shared__ float s_aw[T_ENC_ + 30];
  __shared__ float s_awc[T_ENC_ + 30];
  __shared__ float s_conv[FILT_ * 2 * KS_];
  __shared__ float s_dense[ATTN_ * FILT_];
  __shared__ float s_loc[T_ENC_ * FILT_];
  __shared__ float s_pq[ATTN_], s_vv[ATTN_], s_ab[ATTN_];
  for (int i = tid; i < T_ENC_ + 30; i += 256) {
    int t = i - 15;
    bool ok = (t >= 0 && t < T_ENC_);
    s_aw[i]  = ok ? aw[(size_t)b * T_ENC_ + t] : 0.0f;
    s_awc[i] = ok ? awc[(size_t)b * T_ENC_ + t] : 0.0f;
  }
  for (int i = tid; i < FILT_ * 2 * KS_; i += 256) s_conv[i] = convw[i];
  for (int i = tid; i < ATTN_ * FILT_; i += 256) s_dense[i] = densew[i];
  if (tid < ATTN_) {
    s_pq[tid] = pq[(size_t)b * ATTN_ + tid];
    s_vv[tid] = vv[tid];
    s_ab[tid] = abias[tid];
  }
  __syncthreads();
  // conv: loc[t][f]
  for (int i = tid; i < T_ENC_ * FILT_; i += 256) {
    int t = i >> 5;
    int f = i & 31;
    const float* w = s_conv + f * (2 * KS_);
    float s = 0.0f;
#pragma unroll
    for (int k = 0; k < KS_; ++k)
      s += w[k] * s_aw[t + k] + w[KS_ + k] * s_awc[t + k];
    s_loc[t * FILT_ + f] = s;
  }
  __syncthreads();
  int t = tid;  // 256 threads, 256 encoder positions
  float e = 0.0f;
  const float* pmrow = pm + ((size_t)b * T_ENC_ + t) * ATTN_;
  const float* lr = s_loc + t * FILT_;
  for (int a = 0; a < ATTN_; ++a) {
    const float* dw = s_dense + a * FILT_;
    float pl = 0.0f;
#pragma unroll
    for (int f = 0; f < FILT_; ++f) pl += dw[f] * lr[f];
    e += s_vv[a] * tanhf(s_pq[a] + pl + pmrow[a] + s_ab[a]);
  }
  energies[(size_t)b * T_ENC_ + t] = e;
}

// Softmax over T_ENC, update aw/awc, compute context, write alignments.
__global__ void softmax_ctx_kernel(const float* __restrict__ energies,
                                   float* __restrict__ aw,
                                   float* __restrict__ awc,
                                   const float* __restrict__ memory,
                                   float* __restrict__ ctx,
                                   float* __restrict__ align_out,
                                   int t_step) {
  int b = blockIdx.x;
  int tid = threadIdx.x;
  __shared__ float s_a[T_ENC_];
  __shared__ float red[T_ENC_];
  float v = energies[(size_t)b * T_ENC_ + tid];
  red[tid] = v;
  __syncthreads();
  for (int s = 128; s > 0; s >>= 1) {
    if (tid < s) red[tid] = fmaxf(red[tid], red[tid + s]);
    __syncthreads();
  }
  float mx = red[0];
  __syncthreads();
  float ex = __expf(v - mx);
  red[tid] = ex;
  __syncthreads();
  for (int s = 128; s > 0; s >>= 1) {
    if (tid < s) red[tid] += red[tid + s];
    __syncthreads();
  }
  float a = ex / red[0];
  s_a[tid] = a;
  aw[(size_t)b * T_ENC_ + tid] = a;
  awc[(size_t)b * T_ENC_ + tid] += a;
  align_out[((size_t)b * T_DEC_ + t_step) * T_ENC_ + tid] = a;
  __syncthreads();
  for (int e = tid; e < EMB_; e += 256) {
    const float* mp = memory + (size_t)b * T_ENC_ * EMB_ + e;
    float s = 0.0f;
    for (int t = 0; t < T_ENC_; ++t) s += s_a[t] * mp[(size_t)t * EMB_];
    ctx[(size_t)b * EMB_ + e] = s;
  }
}

// mel = [dh | ctx] @ proj_w^T + proj_b
__global__ void proj_kernel(const float* __restrict__ dh,
                            const float* __restrict__ ctx,
                            const float* __restrict__ proj_w,
                            const float* __restrict__ proj_b,
                            float* __restrict__ mel_out, int t_step) {
  int idx = blockIdx.x * blockDim.x + threadIdx.x;
  if (idx >= B_ * MEL_) return;
  int b = idx / MEL_;
  int j = idx % MEL_;
  const float* w = proj_w + (size_t)j * (LSTM_ + EMB_);
  float s = proj_b[j];
  const float* hb = dh + (size_t)b * LSTM_;
  for (int k = 0; k < LSTM_; ++k) s += hb[k] * w[k];
  const float* cb = ctx + (size_t)b * EMB_;
  for (int k = 0; k < EMB_; ++k) s += cb[k] * w[LSTM_ + k];
  mel_out[((size_t)b * T_DEC_ + t_step) * MEL_ + j] = s;
}

// ---------------------------------------------------------------------------
// Host launcher
// ---------------------------------------------------------------------------
extern "C" void kernel_launch(void* const* d_in, const int* in_sizes, int n_in,
                              void* d_out, int out_size, void* d_ws,
                              size_t ws_size, hipStream_t stream) {
  const float* memory     = (const float*)d_in[0];
  const float* dec_inputs = (const float*)d_in[1];
  // d_in[2] memory_lengths: unused by the reference computation
  const float* prenet_w1  = (const float*)d_in[3];
  const float* prenet_w2  = (const float*)d_in[4];
  const float* att_wih    = (const float*)d_in[5];
  const float* att_whh    = (const float*)d_in[6];
  const float* att_bih    = (const float*)d_in[7];
  const float* att_bhh    = (const float*)d_in[8];
  const float* dec_wih    = (const float*)d_in[9];
  const float* dec_whh    = (const float*)d_in[10];
  const float* dec_bih    = (const float*)d_in[11];
  const float* dec_bhh    = (const float*)d_in[12];
  const float* query_w    = (const float*)d_in[13];
  const float* memory_w   = (const float*)d_in[14];
  const float* v_w        = (const float*)d_in[15];
  const float* loc_conv_w = (const float*)d_in[16];
  const float* loc_dense_w= (const float*)d_in[17];
  const float* attn_var   = (const float*)d_in[18];
  const float* attn_bias  = (const float*)d_in[19];
  const float* proj_w     = (const float*)d_in[20];
  const float* proj_b     = (const float*)d_in[21];

  float* mel_out   = (float*)d_out;
  float* align_out = mel_out + (size_t)B_ * T_DEC_ * MEL_;

  // ---- workspace carve-up (256B aligned) ----
  char* ws = (char*)d_ws;
  size_t off = 0;
  auto alloc = [&](size_t bytes) -> char* {
    char* p = ws + off;
    off += (bytes + 255) & ~(size_t)255;
    return p;
  };
  const size_t WCAT = (size_t)(4 * LSTM_) * (2 * LSTM_);
  __bf16* Wcat_att = (__bf16*)alloc(WCAT * 2);
  __bf16* Wcat_dec = (__bf16*)alloc(WCAT * 2);
  float*  bias_att = (float*)alloc(4 * LSTM_ * 4);
  float*  bias_dec = (float*)alloc(4 * LSTM_ * 4);
  __bf16* qw_bf    = (__bf16*)alloc((size_t)ATTN_ * LSTM_ * 2);
  __bf16* mw_bf    = (__bf16*)alloc((size_t)ATTN_ * EMB_ * 2);
  __bf16* w2_bf    = (__bf16*)alloc((size_t)PRE_ * PRE_ * 2);
  const size_t MEMN = (size_t)B_ * T_ENC_ * EMB_;
  __bf16* mem_bf   = (__bf16*)alloc(MEMN * 2);
  float*  pm       = (float*)alloc((size_t)B_ * T_ENC_ * ATTN_ * 4);
  const size_t H1N = (size_t)T_DEC_ * B_ * PRE_;
  float*  h1       = (float*)alloc(H1N * 4);
  __bf16* h1_bf    = (__bf16*)alloc(H1N * 2);
  float*  xs       = (float*)alloc(H1N * 4);
  __bf16* A_att    = (__bf16*)alloc((size_t)B_ * 2 * LSTM_ * 2);
  __bf16* A_dec    = (__bf16*)alloc((size_t)B_ * 2 * LSTM_ * 2);
  __bf16* ah_bf    = (__bf16*)alloc((size_t)B_ * LSTM_ * 2);
  float*  pq       = (float*)alloc((size_t)B_ * ATTN_ * 4);
  float*  energ    = (float*)alloc((size_t)B_ * T_ENC_ * 4);
  float*  vv       = (float*)alloc(ATTN_ * 4);
  // contiguous state block (each piece is a multiple of 256B)
  float* state = (float*)alloc(((size_t)4 * B_ * LSTM_ + B_ * EMB_ +
                                2 * (size_t)B_ * T_ENC_) * 4);
  float* ah  = state;
  float* ac  = ah + (size_t)B_ * LSTM_;
  float* dh  = ac + (size_t)B_ * LSTM_;
  float* dc  = dh + (size_t)B_ * LSTM_;
  float* ctx = dc + (size_t)B_ * LSTM_;
  float* aw  = ctx + (size_t)B_ * EMB_;
  float* awc = aw + (size_t)B_ * T_ENC_;
  const size_t STATE_N =
      (size_t)4 * B_ * LSTM_ + B_ * EMB_ + 2 * (size_t)B_ * T_ENC_;
  (void)ws_size; (void)n_in; (void)in_sizes; (void)out_size;

  // ---- precompute (parallel) ----
  cat_weights_bf16<<<(int)((WCAT + 255) / 256), 256, 0, stream>>>(
      att_wih, att_whh, Wcat_att);
  cat_weights_bf16<<<(int)((WCAT + 255) / 256), 256, 0, stream>>>(
      dec_wih, dec_whh, Wcat_dec);
  bias_add_kernel<<<16, 256, 0, stream>>>(att_bih, att_bhh, bias_att, 4 * LSTM_);
  bias_add_kernel<<<16, 256, 0, stream>>>(dec_bih, dec_bhh, bias_dec, 4 * LSTM_);
  cvt_f32_to_bf16<<<512, 256, 0, stream>>>(query_w, qw_bf,
                                           (size_t)ATTN_ * LSTM_);
  cvt_f32_to_bf16<<<384, 256, 0, stream>>>(memory_w, mw_bf,
                                           (size_t)ATTN_ * EMB_);
  cvt_f32_to_bf16<<<256, 256, 0, stream>>>(prenet_w2, w2_bf,
                                           (size_t)PRE_ * PRE_);
  cvt_f32_to_bf16<<<(int)((MEMN + 255) / 256), 256, 0, stream>>>(memory, mem_bf,
                                                                 MEMN);
  mul_vec_kernel<<<1, 128, 0, stream>>>(attn_var, v_w, vv, ATTN_);
  zero_f32_kernel<<<(int)((STATE_N + 255) / 256), 256, 0, stream>>>(state,
                                                                    STATE_N);

  // processed_memory = memory @ memory_w^T  (M=16384, N=128, K=768)
  {
    dim3 g(ATTN_ / 64, (B_ * T_ENC_) / 64);
    wmma_gemm_bf16<<<g, 128, 0, stream>>>(mem_bf, mw_bf, pm, B_ * T_ENC_,
                                          ATTN_, EMB_, 0);
  }
  // prenet: layer1 VALU (K=80), layer2 WMMA (M=12800, N=256, K=256)
  prenet1_kernel<<<(int)((H1N + 255) / 256), 256, 0, stream>>>(dec_inputs,
                                                               prenet_w1, h1);
  cvt_f32_to_bf16<<<(int)((H1N + 255) / 256), 256, 0, stream>>>(h1, h1_bf, H1N);
  {
    dim3 g(PRE_ / 64, (T_DEC_ * B_) / 64);
    wmma_gemm_bf16<<<g, 128, 0, stream>>>(h1_bf, w2_bf, xs, T_DEC_ * B_, PRE_,
                                          PRE_, 1);
  }

  // ---- sequential decode loop ----
  for (int t = 0; t < T_DEC_; ++t) {
    const float* xs_t = xs + (size_t)t * B_ * PRE_;
    // attention LSTM: A = [x_t | ctx | ah]
    pack_A_kernel<<<512, 256, 0, stream>>>(xs_t, ctx, ah, A_att);
    lstm_fused_kernel<<<LSTM_ / 16, 128, 0, stream>>>(A_att, Wcat_att, bias_att,
                                                      ah, ac);
    // pq = ah @ query_w^T  (M=64, N=128, K=1024)
    cvt_f32_to_bf16<<<256, 256, 0, stream>>>(ah, ah_bf, (size_t)B_ * LSTM_);
    {
      dim3 g(ATTN_ / 64, B_ / 64);
      wmma_gemm_bf16<<<g, 128, 0, stream>>>(ah_bf, qw_bf, pq, B_, ATTN_, LSTM_,
                                            0);
    }
    // energies (location conv + tanh reduce), softmax + context
    energies_kernel<<<B_, 256, 0, stream>>>(aw, awc, pq, pm, loc_conv_w,
                                            loc_dense_w, vv, attn_bias, energ);
    softmax_ctx_kernel<<<B_, T_ENC_, 0, stream>>>(energ, aw, awc, memory, ctx,
                                                  align_out, t);
    // decoder LSTM: A = [x_t | ctx_new | dh]
    pack_A_kernel<<<512, 256, 0, stream>>>(xs_t, ctx, dh, A_dec);
    lstm_fused_kernel<<<LSTM_ / 16, 128, 0, stream>>>(A_dec, Wcat_dec, bias_dec,
                                                      dh, dc);
    // projection
    proj_kernel<<<(B_ * MEL_ + 255) / 256, 256, 0, stream>>>(dh, ctx, proj_w,
                                                             proj_b, mel_out, t);
  }
}
